// Attention_8254927143187
// MI455X (gfx1250) — matrix operand order repaired and checked by reading, not compiled, not run on hardware
//
#include <hip/hip_runtime.h>
#include <hip/hip_bf16.h>
#include <math.h>
#include <stdint.h>

// ---------- types ----------
typedef __attribute__((ext_vector_type(16))) __bf16        v16bf;
typedef __attribute__((ext_vector_type(8)))  float         v8f;
typedef __attribute__((ext_vector_type(4)))  unsigned int  u32x4;
typedef __attribute__((ext_vector_type(4)))  int           i32x4;

union FragB16 {
    v16bf          bf;
    u32x4          u[2];
    unsigned short h[16];
};

__device__ __forceinline__ unsigned short f32_to_bf16_rn(float f) {
    unsigned int u = __float_as_uint(f);
    u += 0x7FFFu + ((u >> 16) & 1u);          // round-to-nearest-even
    return (unsigned short)(u >> 16);
}

// ---------- CDNA5 async global->LDS copy (ASYNCcnt path), with safe fallback ----
#define GLOBAL_AS __attribute__((address_space(1)))
#define LDS_AS    __attribute__((address_space(3)))

__device__ __forceinline__ void async_copy_b128(const unsigned short* g,
                                                unsigned short* l) {
#if __has_builtin(__builtin_amdgcn_global_load_async_to_lds_b128)
    __builtin_amdgcn_global_load_async_to_lds_b128(
        (GLOBAL_AS i32x4*)(uintptr_t)g,
        (LDS_AS i32x4*)(unsigned int)(uintptr_t)l, 0, 0);
#else
    *(u32x4*)l = *(const u32x4*)g;            // global->vgpr->ds_store_b128
#endif
}

#if __has_builtin(__builtin_amdgcn_s_wait_asynccnt)
#define ASYNC_WAIT0() __builtin_amdgcn_s_wait_asynccnt(0)
#else
#define ASYNC_WAIT0() asm volatile("s_wait_asynccnt 0" ::: "memory")
#endif

// ---------- elementwise fp32 -> bf16 ----------
__global__ void f32_to_bf16_kernel(const float* __restrict__ in,
                                   unsigned short* __restrict__ out, int n) {
    int i = blockIdx.x * blockDim.x + threadIdx.x;
    if (i < n) out[i] = f32_to_bf16_rn(in[i]);
}

// ---------- RoPE (+optional softmax scale), [B,S,H,128] fp32 -> bf16 ----------
__global__ void rope_bf16_kernel(const float* __restrict__ in,
                                 unsigned short* __restrict__ out,
                                 int total_pairs, float outscale) {
    int i = blockIdx.x * blockDim.x + threadIdx.x;
    if (i >= total_pairs) return;
    int dp  = i & 63;            // pair index within head dim (0..63)
    int row = i >> 6;            // (b*S + s)*16 + h
    int s   = (row >> 4) & 2047; // S == 2048
    float freq = __expf(-(float)(2 * dp) * (9.210340371976184f / 128.0f));
    float ang  = (float)s * freq;
    float cs = __cosf(ang), sn = __sinf(ang);
    size_t base = (size_t)row * 128 + 2 * dp;
    float xr = in[base], xi = in[base + 1];
    out[base]     = f32_to_bf16_rn((xr * cs - xi * sn) * outscale);
    out[base + 1] = f32_to_bf16_rn((xr * sn + xi * cs) * outscale);
}

// ---------- V: [B,S,H,128] fp32 -> Vt [B,H,128,S] bf16 ----------
__global__ void vtrans_bf16_kernel(const float* __restrict__ in,
                                   unsigned short* __restrict__ out, int total) {
    int i = blockIdx.x * blockDim.x + threadIdx.x;
    if (i >= total) return;
    int d = i & 127;
    int h = (i >> 7) & 15;
    int s = (i >> 11) & 2047;
    int b = i >> 22;
    out[((size_t)((b * 16 + h) * 128 + d)) * 2048 + s] = f32_to_bf16_rn(in[i]);
}

// ---------- GEMM: C[M,N] fp32 = A[M,K]bf16 * (B[N,K])^T bf16 ----------
// grid (N/64, M/128), block 128 (4 waves). Each wave: 32 rows x 64 cols.
// B tile (64x32) shared block-wide, double-buffered in LDS via async copies;
// A fragments register double-buffered with COMPILE-TIME buffer indices
// (2x-unrolled k loop) so nothing spills to scratch.
__global__ __launch_bounds__(128)
void gemm_bf16_nt(const unsigned short* __restrict__ A,
                  const unsigned short* __restrict__ Bm,
                  float* __restrict__ C, int M, int N, int K) {
    __shared__ __align__(16) unsigned short btile[2][64 * 32];
    const int tid  = threadIdx.x;
    const int lane = tid & 31;
    const int wave = tid >> 5;
    const int lm   = lane & 15;
    const int hi   = lane >> 4;
    const int m0   = blockIdx.y * 128 + wave * 32;
    const int n0   = blockIdx.x * 64;

    const unsigned short* Arow0 = A + (size_t)(m0 + lm) * K;
    const unsigned short* Arow1 = A + (size_t)(m0 + 16 + lm) * K;

    // cooperative async fill of one 64x32 B tile (4 KB): 2x b128 per thread
    auto fillB = [&](int buf, int k0) {
#pragma unroll
        for (int p = 0; p < 2; p++) {
            int c   = tid + p * 128;      // 0..255 16B-chunks
            int row = c >> 2;             // n within tile (0..63)
            int q   = c & 3;              // 16B quarter of the 64B row
            async_copy_b128(Bm + (size_t)(n0 + row) * K + k0 + q * 8,
                            &btile[buf][row * 32 + q * 8]);
        }
    };

    v8f     acc[8] = {};                  // [mi*4 + t]
    FragB16 a0[2], a1[2];                 // two named reg buffers (no spill)

    fillB(0, 0);
    a0[0].u[0] = *(const u32x4*)(Arow0 + hi * 8);
    a0[0].u[1] = *(const u32x4*)(Arow0 + hi * 8 + 16);
    a0[1].u[0] = *(const u32x4*)(Arow1 + hi * 8);
    a0[1].u[1] = *(const u32x4*)(Arow1 + hi * 8 + 16);
    ASYNC_WAIT0();
    __syncthreads();

    const int nsteps = K / 32;            // even (K multiple of 64)

#define GEMM_STEP(CURA, NXTA, CURBUF, NXTBUF, KK)                             \
    {                                                                         \
        const int k0s = (KK) * 32;                                            \
        if ((KK) + 1 < nsteps) {                                              \
            fillB((NXTBUF), k0s + 32);                                        \
            const int ka = k0s + 32 + hi * 8;                                 \
            NXTA[0].u[0] = *(const u32x4*)(Arow0 + ka);                       \
            NXTA[0].u[1] = *(const u32x4*)(Arow0 + ka + 16);                  \
            NXTA[1].u[0] = *(const u32x4*)(Arow1 + ka);                       \
            NXTA[1].u[1] = *(const u32x4*)(Arow1 + ka + 16);                  \
        }                                                                     \
        _Pragma("unroll")                                                     \
        for (int t = 0; t < 4; t++) {                                         \
            FragB16 bfr;                                                      \
            const unsigned short* bp =                                        \
                &btile[(CURBUF)][(t * 16 + lm) * 32 + hi * 8];                \
            bfr.u[0] = *(const u32x4*)(bp);                                   \
            bfr.u[1] = *(const u32x4*)(bp + 16);                              \
            acc[t] = __builtin_amdgcn_wmma_f32_16x16x32_bf16(                 \
                false, CURA[0].bf, false, bfr.bf, (short)0, acc[t],           \
                false, false);                                                \
            acc[4 + t] = __builtin_amdgcn_wmma_f32_16x16x32_bf16(             \
                false, CURA[1].bf, false, bfr.bf, (short)0, acc[4 + t],       \
                false, false);                                                \
        }                                                                     \
        ASYNC_WAIT0();                                                        \
        __syncthreads();                                                      \
    }

    for (int kk = 0; kk < nsteps; kk += 2) {
        GEMM_STEP(a0, a1, 0, 1, kk);
        GEMM_STEP(a1, a0, 1, 0, kk + 1);
    }
#undef GEMM_STEP

#pragma unroll
    for (int mi = 0; mi < 2; mi++)
#pragma unroll
        for (int t = 0; t < 4; t++) {
            int n = n0 + t * 16 + lm;
#pragma unroll
            for (int r = 0; r < 8; r++) {
                int m = m0 + mi * 16 + r + hi * 8;
                C[(size_t)m * N + n] = acc[mi * 4 + t][r];
            }
        }
}

// ---------- Flash attention, one wave = 16 q-rows, 32-key tiles ----------
// Qr,Kr: [B,S,H,128] bf16 (Q pre-scaled by 1/sqrt(128), both RoPE'd)
// Vt: [B,H,128,S] bf16.  AO: [B,S,H,128] bf16.
__global__ __launch_bounds__(128)
void attn_kernel(const unsigned short* __restrict__ Qr,
                 const unsigned short* __restrict__ Kr,
                 const unsigned short* __restrict__ Vt,
                 unsigned short* __restrict__ AO) {
    __shared__ __align__(16) unsigned short plds[4][16 * 32];
    const int lane  = threadIdx.x & 31;
    const int wave  = threadIdx.x >> 5;
    const int lm    = lane & 15;
    const int hi    = lane >> 4;
    const int h     = blockIdx.y;
    const int b     = blockIdx.z;
    const int qbase = blockIdx.x * 64 + wave * 16;

    FragB16 qf[4];
    const unsigned short* Qrow =
        Qr + ((size_t)b * 2048 + qbase + lm) * 2048 + h * 128;
#pragma unroll
    for (int c = 0; c < 4; c++) {
        qf[c].u[0] = *(const u32x4*)(Qrow + c * 32 + hi * 8);
        qf[c].u[1] = *(const u32x4*)(Qrow + c * 32 + hi * 8 + 16);
    }

    v8f o[8] = {};
    float rmax[8], rsum[8];
#pragma unroll
    for (int r = 0; r < 8; r++) { rmax[r] = -1e30f; rsum[r] = 0.0f; }

    const unsigned short* Kbase = Kr + (size_t)b * 2048 * 2048 + h * 128;
    const unsigned short* Vbase = Vt + ((size_t)(b * 16 + h)) * 128 * 2048;

    for (int j0 = 0; j0 < 2048; j0 += 32) {
        v8f s0 = {}, s1 = {};
#pragma unroll
        for (int c = 0; c < 4; c++) {
            FragB16 kf0, kf1;
            const unsigned short* kp0 =
                Kbase + (size_t)(j0 + lm) * 2048 + c * 32 + hi * 8;
            const unsigned short* kp1 =
                Kbase + (size_t)(j0 + 16 + lm) * 2048 + c * 32 + hi * 8;
            kf0.u[0] = *(const u32x4*)(kp0);
            kf0.u[1] = *(const u32x4*)(kp0 + 16);
            kf1.u[0] = *(const u32x4*)(kp1);
            kf1.u[1] = *(const u32x4*)(kp1 + 16);
            s0 = __builtin_amdgcn_wmma_f32_16x16x32_bf16(
                false, qf[c].bf, false, kf0.bf, (short)0, s0, false, false);
            s1 = __builtin_amdgcn_wmma_f32_16x16x32_bf16(
                false, qf[c].bf, false, kf1.bf, (short)0, s1, false, false);
        }
        float sc[8];
#pragma unroll
        for (int r = 0; r < 8; r++) {
            float mx = fmaxf(s0[r], s1[r]);
            mx = fmaxf(mx, __shfl_xor(mx, 1, 32));
            mx = fmaxf(mx, __shfl_xor(mx, 2, 32));
            mx = fmaxf(mx, __shfl_xor(mx, 4, 32));
            mx = fmaxf(mx, __shfl_xor(mx, 8, 32));
            float nm = fmaxf(rmax[r], mx);
            sc[r] = __expf(rmax[r] - nm);
            float e0 = __expf(s0[r] - nm);
            float e1 = __expf(s1[r] - nm);
            float rs = e0 + e1;
            rs += __shfl_xor(rs, 1, 32);
            rs += __shfl_xor(rs, 2, 32);
            rs += __shfl_xor(rs, 4, 32);
            rs += __shfl_xor(rs, 8, 32);
            rsum[r] = rsum[r] * sc[r] + rs;
            rmax[r] = nm;
            int row = r + hi * 8;
            plds[wave][row * 32 + lm]      = f32_to_bf16_rn(e0);
            plds[wave][row * 32 + 16 + lm] = f32_to_bf16_rn(e1);
        }
#pragma unroll
        for (int f = 0; f < 8; f++)
#pragma unroll
            for (int r = 0; r < 8; r++) o[f][r] *= sc[r];

        asm volatile("s_wait_dscnt 0" ::: "memory");
        FragB16 pf;
        const unsigned short* prow = &plds[wave][lm * 32] + hi * 8;
        pf.u[0] = *(const u32x4*)(prow);
        pf.u[1] = *(const u32x4*)(prow + 16);
        asm volatile("" ::: "memory");

#pragma unroll
        for (int f = 0; f < 8; f++) {
            FragB16 vb;
            const unsigned short* vp =
                Vbase + (size_t)(f * 16 + lm) * 2048 + j0 + hi * 8;
            vb.u[0] = *(const u32x4*)(vp);
            vb.u[1] = *(const u32x4*)(vp + 16);
            o[f] = __builtin_amdgcn_wmma_f32_16x16x32_bf16(
                false, pf.bf, false, vb.bf, (short)0, o[f], false, false);
        }
    }

    float inv[8];
#pragma unroll
    for (int r = 0; r < 8; r++) inv[r] = 1.0f / rsum[r];
#pragma unroll
    for (int f = 0; f < 8; f++)
#pragma unroll
        for (int r = 0; r < 8; r++) {
            size_t row = (size_t)b * 2048 + qbase + r + hi * 8;
            AO[row * 2048 + h * 128 + f * 16 + lm] =
                f32_to_bf16_rn(o[f][r] * inv[r]);
        }
}

// ---------- host launcher ----------
extern "C" void kernel_launch(void* const* d_in, const int* in_sizes, int n_in,
                              void* d_out, int out_size, void* d_ws, size_t ws_size,
                              hipStream_t stream) {
    const float* x  = (const float*)d_in[0];
    const float* wq = (const float*)d_in[1];
    const float* wk = (const float*)d_in[2];
    const float* wv = (const float*)d_in[3];
    const float* wo = (const float*)d_in[4];
    float* out = (float*)d_out;

    constexpr int    Bz = 2, S = 2048, D = 2048, H = 16;
    constexpr int    M  = Bz * S;                 // 4096
    constexpr size_t NX = (size_t)M * D;          // 8,388,608
    constexpr size_t NW = (size_t)D * D;          // 4,194,304
    constexpr int    PAIRS = M * H * 64;          // 4,194,304

    char* ws = (char*)d_ws;
    size_t off = 0;
    auto take = [&](size_t bytes) -> char* {
        char* p = ws + off;
        off += (bytes + 255) & ~(size_t)255;
        return p;
    };
    unsigned short* x_bf  = (unsigned short*)take(NX * 2);
    unsigned short* wq_bf = (unsigned short*)take(NW * 2);
    unsigned short* wk_bf = (unsigned short*)take(NW * 2);
    unsigned short* wv_bf = (unsigned short*)take(NW * 2);
    unsigned short* wo_bf = (unsigned short*)take(NW * 2);
    float*          tmp   = (float*)take(NX * 4);
    unsigned short* Qr    = (unsigned short*)take(NX * 2);
    unsigned short* Kr    = (unsigned short*)take(NX * 2);
    unsigned short* Vt    = (unsigned short*)take(NX * 2);
    unsigned short* AO    = x_bf;   // x_bf dead after the QKV GEMMs

    const int T = 256;
    f32_to_bf16_kernel<<<(int)((NX + T - 1) / T), T, 0, stream>>>(x, x_bf, (int)NX);
    f32_to_bf16_kernel<<<(int)((NW + T - 1) / T), T, 0, stream>>>(wq, wq_bf, (int)NW);
    f32_to_bf16_kernel<<<(int)((NW + T - 1) / T), T, 0, stream>>>(wk, wk_bf, (int)NW);
    f32_to_bf16_kernel<<<(int)((NW + T - 1) / T), T, 0, stream>>>(wv, wv_bf, (int)NW);
    f32_to_bf16_kernel<<<(int)((NW + T - 1) / T), T, 0, stream>>>(wo, wo_bf, (int)NW);

    dim3 ggrid(D / 64, M / 128);   // (32, 32)

    gemm_bf16_nt<<<ggrid, 128, 0, stream>>>(x_bf, wq_bf, tmp, M, D, D);
    rope_bf16_kernel<<<(PAIRS + T - 1) / T, T, 0, stream>>>(
        tmp, Qr, PAIRS, 0.08838834764831845f);
    gemm_bf16_nt<<<ggrid, 128, 0, stream>>>(x_bf, wk_bf, tmp, M, D, D);
    rope_bf16_kernel<<<(PAIRS + T - 1) / T, T, 0, stream>>>(tmp, Kr, PAIRS, 1.0f);
    gemm_bf16_nt<<<ggrid, 128, 0, stream>>>(x_bf, wv_bf, tmp, M, D, D);
    vtrans_bf16_kernel<<<(int)((NX + T - 1) / T), T, 0, stream>>>(tmp, Vt, (int)NX);

    attn_kernel<<<dim3(S / 64, H, Bz), 128, 0, stream>>>(Qr, Kr, Vt, AO);

    gemm_bf16_nt<<<ggrid, 128, 0, stream>>>(AO, wo_bf, out, M, D, D);
}
